// EdgePredictor_12335146074240
// MI455X (gfx1250) — matrix operand order repaired and checked by reading, not compiled, not run on hardware
//
#include <hip/hip_runtime.h>
#include <hip/hip_bf16.h>
#include <stdint.h>

typedef __attribute__((ext_vector_type(16))) __bf16 v16bf;
typedef __attribute__((ext_vector_type(8)))  float  v8f;

#define D_NODE 128
#define EFEAT  32
#define IN_DIM 288
#define HID    256
#define HID2   128
#define BM     64              // edges per block
#define MT     (BM/16)         // 4 m-tiles
#define KC1    (IN_DIM/32)     // 9 K-chunks layer1
#define KC2    (HID/32)        // 8 K-chunks layer2
#define KC3    (HID2/32)       // 4 K-chunks layer3
#define NT1    (HID/16)        // 16 n-tiles layer1
#define NT2    (HID2/16)       // 8 n-tiles layer2

// bf16 weight fragments in workspace (element offsets)
#define W1F_OFF   0
#define W1F_ELEMS (NT1*KC1*512)   // 73728
#define W2F_OFF   (W1F_OFF + W1F_ELEMS)
#define W2F_ELEMS (NT2*KC2*512)   // 32768
#define W3F_OFF   (W2F_OFF + W2F_ELEMS)
#define W3F_ELEMS (KC3*512)       // 2048 (N padded 2 -> 16 with zeros)

__device__ __forceinline__ unsigned f2bf(float f) {
    union { float f; unsigned u; } x; x.f = f;
    unsigned r = x.u + 0x7FFFu + ((x.u >> 16) & 1u);   // round-to-nearest-even
    return r >> 16;
}

// A-matrix fragment element index (16-bit A, 16x32): lane = half*16 + m,
// element e covers K = (e&7) + ((e>>3)<<4) + half*8
__device__ __forceinline__ int a_elem_off(int frag, int m, int kk) {
    int half = (kk >> 3) & 1;
    int e    = (kk & 7) | ((kk & 16) ? 8 : 0);
    return (frag * 32 + (half * 16 + m)) * 16 + e;
}

__device__ __forceinline__ v8f wmma_bf16(v16bf a, v16bf b, v8f c) {
    return __builtin_amdgcn_wmma_f32_16x16x32_bf16(false, a, false, b,
                                                   (short)0, c, false, false);
}

// ---------------- weight pre-pack: f32 -> bf16, B-fragment order ----------------
// B-matrix (16-bit, 32x16): lane l, element e -> K = (e&7) + ((e>>3)<<4) + (l>>4)*8, N = l&15
__global__ void pack_weights(const float* __restrict__ w1,
                             const float* __restrict__ w2,
                             const float* __restrict__ w3,
                             unsigned short* __restrict__ ws) {
    int i = blockIdx.x * 256 + threadIdx.x;
    if (i < W1F_ELEMS) {
        int frag = i >> 9, w = i & 511;
        int l = w >> 4, e = w & 15;
        int n = frag / KC1, c = frag % KC1;
        int k = c * 32 + ((l >> 4) << 3) + (e & 7) + ((e >> 3) << 4);
        ws[W1F_OFF + i] = (unsigned short)f2bf(w1[k * HID + n * 16 + (l & 15)]);
    } else if (i < W1F_ELEMS + W2F_ELEMS) {
        int j = i - W1F_ELEMS;
        int frag = j >> 9, w = j & 511;
        int l = w >> 4, e = w & 15;
        int n = frag / KC2, c = frag % KC2;
        int k = c * 32 + ((l >> 4) << 3) + (e & 7) + ((e >> 3) << 4);
        ws[W2F_OFF + j] = (unsigned short)f2bf(w2[k * HID2 + n * 16 + (l & 15)]);
    } else if (i < W1F_ELEMS + W2F_ELEMS + W3F_ELEMS) {
        int j = i - W1F_ELEMS - W2F_ELEMS;
        int c = j >> 9, w = j & 511;
        int l = w >> 4, e = w & 15;
        int k = c * 32 + ((l >> 4) << 3) + (e & 7) + ((e >> 3) << 4);
        int nn = l & 15;
        ws[W3F_OFF + j] = (nn < 2) ? (unsigned short)f2bf(w3[k * 2 + nn])
                                   : (unsigned short)0;
    }
}

// ---------------- fused gather + 3-layer MLP ----------------
__global__ __launch_bounds__(256) void edge_mlp(
    const float* __restrict__ nodes, const long long* __restrict__ eidx,
    const float* __restrict__ ef,
    const float* __restrict__ b1, const float* __restrict__ b2,
    const float* __restrict__ b3,
    const unsigned short* __restrict__ ws,
    float* __restrict__ out, long long n_edges)
{
    __shared__ __attribute__((aligned(32))) unsigned short sA [MT * KC1 * 512]; // 36KB
    __shared__ __attribute__((aligned(32))) unsigned short sH1[MT * KC2 * 512]; // 32KB
    __shared__ __attribute__((aligned(32))) unsigned short sH2[MT * KC3 * 512]; // 16KB

    const int tid  = threadIdx.x;
    const int lane = tid & 31;
    const int wave = tid >> 5;
    const long long base = (long long)blockIdx.x * BM;

    // ---- phase 1: gather src|dst|ef, convert to bf16, pack as A fragments ----
    // 4 threads per edge, float4 loads, packed ds_store_b64
    {
        int m = tid >> 2;          // edge within block, 0..63
        int r = tid & 3;           // quarter of the 288-dim row
        long long eg = base + m; if (eg > n_edges - 1) eg = n_edges - 1;
        long long s = eidx[eg];
        long long d = eidx[n_edges + eg];
        const float* srow = nodes + s * D_NODE;
        const float* drow = nodes + d * D_NODE;
        const float* erow = ef + eg * EFEAT;
        int t = m >> 4, mi = m & 15;
        #pragma unroll 6
        for (int q = 0; q < 18; ++q) {
            int k = r * 72 + q * 4;               // segments stay 16B aligned
            float4 v = (k < D_NODE)     ? *(const float4*)(srow + k)
                     : (k < 2 * D_NODE) ? *(const float4*)(drow + k - D_NODE)
                                        : *(const float4*)(erow + k - 2 * D_NODE);
            int c = k >> 5, kk = k & 31;
            int half = (kk >> 3) & 1;
            int e = (kk & 7) | ((kk & 16) ? 8 : 0);            // e % 4 == 0
            int off = ((t * KC1 + c) * 32 + (half * 16 + mi)) * 16 + e;
            uint2 p;
            p.x = f2bf(v.x) | (f2bf(v.y) << 16);
            p.y = f2bf(v.z) | (f2bf(v.w) << 16);
            *(uint2*)&sA[off] = p;                              // 8B aligned
        }
    }
    __syncthreads();

    // ---- layer 1: [64x288] @ [288x256]; each wave owns TWO n-tiles processed
    //      in one pass so every A fragment is read once and register pressure
    //      stays bounded (no loop for LICM to hoist the whole A tile) ----
    {
        const int n0 = wave;           // n-tiles: wave and wave+8
        v8f acc0[MT] = {};
        v8f acc1[MT] = {};
        #pragma unroll
        for (int c = 0; c < KC1; ++c) {
            v16bf b0 = *(const v16bf*)&ws[W1F_OFF + ((n0 * KC1 + c) * 32 + lane) * 16];
            v16bf b1v = *(const v16bf*)&ws[W1F_OFF + (((n0 + 8) * KC1 + c) * 32 + lane) * 16];
            #pragma unroll
            for (int t = 0; t < MT; ++t) {
                v16bf a = *(const v16bf*)&sA[((t * KC1 + c) * 32 + lane) * 16];
                acc0[t] = wmma_bf16(a, b0, acc0[t]);
                acc1[t] = wmma_bf16(a, b1v, acc1[t]);
            }
        }
        #pragma unroll
        for (int h = 0; h < 2; ++h) {
            int n = n0 + h * 8;
            float bias = b1[n * 16 + (lane & 15)];
            int k_ = n * 16 + (lane & 15);
            #pragma unroll
            for (int t = 0; t < MT; ++t)
                #pragma unroll
                for (int v = 0; v < 8; ++v) {
                    float hv = fmaxf((h ? acc1[t][v] : acc0[t][v]) + bias, 0.0f);
                    int m_ = ((lane >> 4) << 3) + v;   // D: M = v + 8*(lane/16)
                    sH1[a_elem_off(t * KC2 + (k_ >> 5), m_, k_ & 31)] =
                        (unsigned short)f2bf(hv);
                }
        }
    }
    __syncthreads();

    // ---- layer 2: [64x256] @ [256x128]; one n-tile per wave ----
    {
        const int n = wave;
        v8f acc[MT] = {};
        #pragma unroll
        for (int c = 0; c < KC2; ++c) {
            v16bf b = *(const v16bf*)&ws[W2F_OFF + ((n * KC2 + c) * 32 + lane) * 16];
            #pragma unroll
            for (int t = 0; t < MT; ++t) {
                v16bf a = *(const v16bf*)&sH1[((t * KC2 + c) * 32 + lane) * 16];
                acc[t] = wmma_bf16(a, b, acc[t]);
            }
        }
        float bias = b2[n * 16 + (lane & 15)];
        int k_ = n * 16 + (lane & 15);
        #pragma unroll
        for (int t = 0; t < MT; ++t)
            #pragma unroll
            for (int v = 0; v < 8; ++v) {
                float h = fmaxf(acc[t][v] + bias, 0.0f);
                int m_ = ((lane >> 4) << 3) + v;
                sH2[a_elem_off(t * KC3 + (k_ >> 5), m_, k_ & 31)] =
                    (unsigned short)f2bf(h);
            }
    }
    __syncthreads();

    // ---- layer 3: [64x128] @ [128x16(pad)] + bias, masked f32 store ----
    if (wave < MT) {
        int t = wave;
        v8f acc = {};
        #pragma unroll
        for (int c = 0; c < KC3; ++c) {
            v16bf a = *(const v16bf*)&sH2[((t * KC3 + c) * 32 + lane) * 16];
            v16bf b = *(const v16bf*)&ws[W3F_OFF + (c * 32 + lane) * 16];
            acc = wmma_bf16(a, b, acc);
        }
        int nn = lane & 15;
        if (nn < 2) {
            float bias = b3[nn];
            #pragma unroll
            for (int v = 0; v < 8; ++v) {
                long long eg = base + t * 16 + ((lane >> 4) << 3) + v;
                if (eg < n_edges) out[eg * 2 + nn] = acc[v] + bias;
            }
        }
    }
}

extern "C" void kernel_launch(void* const* d_in, const int* in_sizes, int n_in,
                              void* d_out, int out_size, void* d_ws, size_t ws_size,
                              hipStream_t stream) {
    const float*     nodes = (const float*)d_in[0];
    const long long* eidx  = (const long long*)d_in[1];
    const float*     ef    = (const float*)d_in[2];
    const float*     w1    = (const float*)d_in[3];
    const float*     b1    = (const float*)d_in[4];
    const float*     w2    = (const float*)d_in[5];
    const float*     b2    = (const float*)d_in[6];
    const float*     w3    = (const float*)d_in[7];
    const float*     b3    = (const float*)d_in[8];
    unsigned short*  wsb   = (unsigned short*)d_ws;
    long long n_edges = (long long)(in_sizes[1] / 2);

    int totalW = W1F_ELEMS + W2F_ELEMS + W3F_ELEMS;
    pack_weights<<<(totalW + 255) / 256, 256, 0, stream>>>(w1, w2, w3, wsb);

    long long nblk = (n_edges + BM - 1) / BM;
    edge_mlp<<<dim3((unsigned)nblk), 256, 0, stream>>>(
        nodes, eidx, ef, b1, b2, b3, wsb, (float*)d_out, n_edges);
}
